// VCTreePredictor_im_68040871903609
// MI455X (gfx1250) — matrix operand order repaired and checked by reading, not compiled, not run on hardware
//
#include <hip/hip_runtime.h>
#include <hip/hip_bf16.h>

#define N_OBJ     1280
#define N_REL     32768
#define HID       512
#define POOL      4096
#define N_OBJ_CLS 151
#define N_REL_CLS 51
#define MT        4   // 16-row tiles per block => M_tile = 64

typedef __attribute__((ext_vector_type(16))) __bf16 v16bf;
typedef __attribute__((ext_vector_type(8)))  __bf16 v8bf;
typedef __attribute__((ext_vector_type(8)))  float  v8f;

// ---------------------------------------------------------------------------
// Pack w_post_cat (1024x4096 f32) into bf16 B-fragment order:
//   w2pk[(kb*POOL + n)*32 + t] = w_post_cat[kb*32 + t][n],  t = 16*halfK + e
// so a lane (halfK = lane/16, n = n0 + lane%16) reads its 16 bf16 contiguously.
// ---------------------------------------------------------------------------
__global__ void pack_w2_kernel(const float* __restrict__ w, __bf16* __restrict__ wp) {
    long long gid = (long long)blockIdx.x * 256 + threadIdx.x;  // 2*HID*POOL total
    if (gid >= (long long)(2 * HID) * POOL) return;
    int t        = (int)(gid & 31);
    long long r  = gid >> 5;
    int n        = (int)(r % POOL);
    int kb       = (int)(r / POOL);
    wp[gid] = (__bf16)w[(long long)(kb * 32 + t) * POOL + n];
}

// ---------------------------------------------------------------------------
// Pack w_ctx (4096x51 f32) into bf16 B-fragment order, N padded 51 -> 64:
//   w3pk[(kb*64 + c)*32 + t] = (c < 51) ? w_ctx[kb*32 + t][c] : 0
// ---------------------------------------------------------------------------
__global__ void pack_w3_kernel(const float* __restrict__ w, __bf16* __restrict__ wp) {
    int gid = blockIdx.x * 256 + threadIdx.x;  // 128*64*32 total
    if (gid >= 128 * 64 * 32) return;
    int t  = gid & 31;
    int r  = gid >> 5;
    int c  = r & 63;
    int kb = r >> 6;
    float v = (c < N_REL_CLS) ? w[(kb * 32 + t) * N_REL_CLS + c] : 0.0f;
    wp[gid] = (__bf16)v;
}

// ---------------------------------------------------------------------------
// GEMM1: edge_rep = relu(edge_ctx @ w_post_emb + b); split into bf16 head/tail.
// 1280x512x1024 = 1.3 GFLOP -> plain f32 VALU is fine (L2-resident weights).
// ---------------------------------------------------------------------------
__global__ void gemm1_relu_split(const float* __restrict__ x, const float* __restrict__ w,
                                 const float* __restrict__ b,
                                 __bf16* __restrict__ head, __bf16* __restrict__ tail) {
    int col = blockIdx.x * 256 + threadIdx.x;  // 0..1023
    int row = blockIdx.y;                      // 0..1279
    float acc = b[col];
    const float* xr = x + (long long)row * HID;
    for (int k = 0; k < HID; ++k)
        acc = fmaf(xr[k], w[(long long)k * (2 * HID) + col], acc);
    acc = fmaxf(acc, 0.0f);
    if (col < HID) head[(long long)row * HID + col]         = (__bf16)acc;
    else           tail[(long long)row * HID + (col - HID)] = (__bf16)acc;
}

// ---------------------------------------------------------------------------
// Fused: gather 64 A rows -> GEMM2 (WMMA bf16, 4x2 register tile per wave)
// -> +bias -> * union -> GEMM3 (WMMA bf16) -> +b_ctx + freq gather -> out.
// One block = 64 relations; 4 waves split the N=4096 dimension; GEMM3
// partials reduced through LDS (overlapped with the dead A-tile region).
// ---------------------------------------------------------------------------
__global__ __launch_bounds__(128)
void fused_rel_kernel(const __bf16* __restrict__ head, const __bf16* __restrict__ tail,
                      const float* __restrict__ unionf,
                      const __bf16* __restrict__ w2pk, const float* __restrict__ b2,
                      const __bf16* __restrict__ w3pk, const float* __restrict__ bctx,
                      const float* __restrict__ freq,
                      const int* __restrict__ obj_preds, const int* __restrict__ pair_idx,
                      float* __restrict__ out) {
    // 128 KB region: A-tile (64x1024 bf16) during the main loop, then reused
    // as the 4x64x64 f32 cross-wave reduction buffer (guarded by barriers).
    __shared__ __align__(32) unsigned char lds_main[16 * MT * 1024 * 2];
    __bf16 (*Atile)[1024] = (__bf16 (*)[1024])lds_main;
    float*  Redf          = (float*)lds_main;              // [4][64][64]
    __shared__ __align__(16) __bf16 Astage[4][MT][16][32]; // 16 KB C->A3 transpose

    const int tid  = threadIdx.x;
    const int wave = tid >> 5;
    const int lane = tid & 31;
    const int half = lane >> 4;   // fragment layouts split at lane 16
    const int l16  = lane & 15;
    const int m0   = blockIdx.x * (16 * MT);

    // ---- stage gathered A tile: row r = concat(head[p0[r]], tail[p1[r]]) ----
    for (int c = tid; c < (16 * MT) * 128; c += 128) {
        int r   = c >> 7;
        int seg = c & 127;                  // 128 x 8-bf16 chunks per row
        const int p = pair_idx[(m0 + r) * 2 + (seg < 64 ? 0 : 1)];
        const __bf16* src = (seg < 64)
            ? (head + (long long)p * HID + seg * 8)
            : (tail + (long long)p * HID + (seg - 64) * 8);
        *(uint4*)&Atile[r][seg * 8] = *(const uint4*)src;
    }
    __syncthreads();

    v8f acc3[MT][4];
    #pragma unroll
    for (int mt = 0; mt < MT; ++mt)
        #pragma unroll
        for (int ct = 0; ct < 4; ++ct)
            #pragma unroll
            for (int j = 0; j < 8; ++j) acc3[mt][ct][j] = 0.0f;

    for (int nc = wave; nc < POOL / 32; nc += 4) {  // 32-wide N chunks of GEMM2
        const int n0 = nc * 32;

        // prefetch next chunk's union_features stream (HBM-resident operand)
        if (nc + 4 < POOL / 32) {
            #pragma unroll
            for (int mt = 0; mt < MT; ++mt)
                __builtin_prefetch(unionf + (long long)(m0 + mt * 16 + half * 8) * POOL + n0 + 128 + l16, 0, 1);
        }

        v8f acc2[MT][2];
        #pragma unroll
        for (int mt = 0; mt < MT; ++mt)
            #pragma unroll
            for (int nt = 0; nt < 2; ++nt)
                #pragma unroll
                for (int j = 0; j < 8; ++j) acc2[mt][nt][j] = 0.0f;

        for (int kb = 0; kb < (2 * HID) / 32; ++kb) {  // K = 1024 in steps of 32
            const int k0 = kb * 32;
            // B fragments: contiguous 32 B per lane from packed layout
            const v16bf b0 = *(const v16bf*)(w2pk + (((long long)kb * POOL + n0 + l16) << 5) + half * 16);
            const v16bf b1 = *(const v16bf*)(w2pk + (((long long)kb * POOL + n0 + 16 + l16) << 5) + half * 16);
            // A fragments (16x32 bf16): K=16*half+e
            #pragma unroll
            for (int mt = 0; mt < MT; ++mt) {
                union { v16bf v; v8bf h[2]; } a;
                a.h[0] = *(const v8bf*)&Atile[mt * 16 + l16][k0 + half * 16];
                a.h[1] = *(const v8bf*)&Atile[mt * 16 + l16][k0 + half * 16 + 8];
                acc2[mt][0] = __builtin_amdgcn_wmma_f32_16x16x32_bf16(false, a.v, false, b0, (short)0, acc2[mt][0], false, false);
                acc2[mt][1] = __builtin_amdgcn_wmma_f32_16x16x32_bf16(false, a.v, false, b1, (short)0, acc2[mt][1], false, false);
            }
        }

        // C layout: VGPR j holds row (j + 8*half), col l16. Apply bias, * union,
        // repack to bf16 through LDS to form the GEMM3 A fragments (transpose).
        const float bias0 = b2[n0 + l16];
        const float bias1 = b2[n0 + 16 + l16];
        #pragma unroll
        for (int mt = 0; mt < MT; ++mt) {
            #pragma unroll
            for (int j = 0; j < 8; ++j) {
                const int cm = j + half * 8;
                const float* urow = unionf + (long long)(m0 + mt * 16 + cm) * POOL + n0;
                float p0 = (acc2[mt][0][j] + bias0) * urow[l16];
                float p1 = (acc2[mt][1][j] + bias1) * urow[16 + l16];
                Astage[wave][mt][cm][l16]      = (__bf16)p0;
                Astage[wave][mt][cm][16 + l16] = (__bf16)p1;
            }
        }
        // same-wave DS store->load; drain DScnt and fence compiler reordering
        asm volatile("s_wait_dscnt 0" ::: "memory");

        union { v16bf v; v8bf h[2]; } a3[MT];
        #pragma unroll
        for (int mt = 0; mt < MT; ++mt) {
            a3[mt].h[0] = *(const v8bf*)&Astage[wave][mt][l16][half * 16];
            a3[mt].h[1] = *(const v8bf*)&Astage[wave][mt][l16][half * 16 + 8];
        }
        #pragma unroll
        for (int ct = 0; ct < 4; ++ct) {  // one B3 fragment shared by all row-tiles
            const v16bf b3 = *(const v16bf*)(w3pk + (((long long)nc * 64 + ct * 16 + l16) << 5) + half * 16);
            #pragma unroll
            for (int mt = 0; mt < MT; ++mt)
                acc3[mt][ct] = __builtin_amdgcn_wmma_f32_16x16x32_bf16(false, a3[mt].v, false, b3, (short)0, acc3[mt][ct], false, false);
        }
    }

    // ---- cross-wave reduction of the 64x64 GEMM3 partials ----
    __syncthreads();  // all waves done with Atile; safe to reuse region as Redf
    #pragma unroll
    for (int mt = 0; mt < MT; ++mt)
        #pragma unroll
        for (int ct = 0; ct < 4; ++ct)
            #pragma unroll
            for (int j = 0; j < 8; ++j)
                Redf[((wave * 16 * MT) + mt * 16 + j + half * 8) * 64 + ct * 16 + l16] = acc3[mt][ct][j];
    __syncthreads();

    for (int i = tid; i < (16 * MT) * 64; i += 128) {
        int m = i >> 6;
        int c = i & 63;
        if (c >= N_REL_CLS) continue;
        float s = Redf[(0 * 16 * MT + m) * 64 + c] + Redf[(1 * 16 * MT + m) * 64 + c]
                + Redf[(2 * 16 * MT + m) * 64 + c] + Redf[(3 * 16 * MT + m) * 64 + c];
        int rel = m0 + m;
        int pa = pair_idx[rel * 2 + 0];
        int pb = pair_idx[rel * 2 + 1];
        long long fi = (long long)(obj_preds[pa] * N_OBJ_CLS + obj_preds[pb]) * N_REL_CLS + c;
        out[(long long)rel * N_REL_CLS + c] = s + bctx[c] + freq[fi];
    }
}

// ---------------------------------------------------------------------------
extern "C" void kernel_launch(void* const* d_in, const int* in_sizes, int n_in,
                              void* d_out, int out_size, void* d_ws, size_t ws_size,
                              hipStream_t stream) {
    const float* edge_ctx   = (const float*)d_in[0];
    const float* unionf     = (const float*)d_in[1];
    const float* w_post_emb = (const float*)d_in[2];
    const float* b_post_emb = (const float*)d_in[3];
    const float* w_post_cat = (const float*)d_in[4];
    const float* b_post_cat = (const float*)d_in[5];
    const float* w_ctx      = (const float*)d_in[6];
    const float* b_ctx      = (const float*)d_in[7];
    const float* freq       = (const float*)d_in[8];
    const int*   obj_preds  = (const int*)d_in[9];
    const int*   pair_idx   = (const int*)d_in[10];
    float* out = (float*)d_out;

    char* ws = (char*)d_ws;
    __bf16* headb = (__bf16*)ws; ws += (size_t)N_OBJ * HID * 2;        // 1.25 MB
    __bf16* tailb = (__bf16*)ws; ws += (size_t)N_OBJ * HID * 2;        // 1.25 MB
    __bf16* w2pk  = (__bf16*)ws; ws += (size_t)2 * HID * POOL * 2;     // 8 MB
    __bf16* w3pk  = (__bf16*)ws; ws += (size_t)128 * 64 * 32 * 2;      // 0.5 MB

    pack_w2_kernel<<<(2 * HID * POOL + 255) / 256, 256, 0, stream>>>(w_post_cat, w2pk);
    pack_w3_kernel<<<(128 * 64 * 32 + 255) / 256, 256, 0, stream>>>(w_ctx, w3pk);
    gemm1_relu_split<<<dim3((2 * HID) / 256, N_OBJ), 256, 0, stream>>>(
        edge_ctx, w_post_emb, b_post_emb, headb, tailb);
    fused_rel_kernel<<<N_REL / (16 * MT), 128, 0, stream>>>(
        headb, tailb, unionf, w2pk, b_post_cat, w3pk, b_ctx, freq, obj_preds, pair_idx, out);
}
